// PowerFlowSoftSuperNodeGNN_12678743458345
// MI455X (gfx1250) — matrix operand order repaired and checked by reading, not compiled, not run on hardware
//
#include <hip/hip_runtime.h>
#include <hip/hip_bf16.h>

typedef __attribute__((ext_vector_type(16))) _Float16 v16h;
typedef __attribute__((ext_vector_type(8)))  float    v8f;

#define HID 32      // hidden dim
#define XW  36      // x16 row stride in halfs (34 used + 2 pad)
#define KP  96      // padded K for msg/node tiles (3 chunks of 32)
#define LAYERS 3

// ---------------------------------------------------------------------------
// A-fragment assembly from a row-major [16 x rowStrideU*2] f16 LDS tile,
// following the CDNA5 16-bit A 16x32 layout:
//   lanes 0-15 : M=lane,    K = chunk*32 + {0..7, 16..23}
//   lanes16-31 : M=lane-16, K = chunk*32 + {8..15, 24..31}
// ---------------------------------------------------------------------------
__device__ __forceinline__ v16h load_a_frag(const _Float16* tile, int lane,
                                            int chunk, int rowStrideU)
{
  const unsigned* tu = (const unsigned*)tile;
  const int row   = lane & 15;
  const int kbase = chunk * 32 + ((lane >> 4) << 3);   // +8 for lanes 16..31
  const unsigned* rp = tu + row * rowStrideU;
  const int kb2 = kbase >> 1;
  union { unsigned u[8]; v16h h; } f;
  f.u[0] = rp[kb2 + 0];  f.u[1] = rp[kb2 + 1];
  f.u[2] = rp[kb2 + 2];  f.u[3] = rp[kb2 + 3];
  f.u[4] = rp[kb2 + 8];  f.u[5] = rp[kb2 + 9];
  f.u[6] = rp[kb2 + 10]; f.u[7] = rp[kb2 + 11];
  return f.h;
}

// ---------------------------------------------------------------------------
// Pre-swizzle weights into B-fragment order (f16).
// B 32x16 layout: lanes 0-15 K={0..7,16..23}, lanes 16-31 K={8..15,24..31} (matches A).
// ---------------------------------------------------------------------------
__global__ void prep_weights(const float* __restrict__ Wmsg,
                             const float* __restrict__ Wnode,
                             const float* __restrict__ Wgn,
                             _Float16* __restrict__ fmsg,
                             _Float16* __restrict__ fnode,
                             _Float16* __restrict__ fgn)
{
  const int stride = gridDim.x * blockDim.x;
  const int tid    = blockIdx.x * blockDim.x + threadIdx.x;
  for (int idx = tid; idx < LAYERS * 6 * 512; idx += stride) {
    const int j = idx & 15, lane = (idx >> 4) & 31;
    const int f = (idx >> 9) % 6, l = idx / (6 * 512);
    const int c = f >> 1, t = f & 1;
    const int k = c * 32 + ((lane >= 16) ? 8 : 0) + ((j < 8) ? j : j + 8);
    const int n = t * 16 + (lane & 15);
    fmsg[idx]  = (_Float16)((k < 70) ? Wmsg [(l * 70 + k) * HID + n] : 0.0f);
    fnode[idx] = (_Float16)((k < 66) ? Wnode[(l * 66 + k) * HID + n] : 0.0f);
  }
  for (int idx = tid; idx < LAYERS * 4 * 512; idx += stride) {
    const int j = idx & 15, lane = (idx >> 4) & 31;
    const int f = (idx >> 9) % 4, l = idx / (4 * 512);
    const int c = f >> 1, t = f & 1;
    const int k = c * 32 + ((lane >= 16) ? 8 : 0) + ((j < 8) ? j : j + 8);
    const int n = t * 16 + (lane & 15);
    fgn[idx] = (_Float16)Wgn[(l * 64 + k) * HID + n];
  }
}

// ---------------------------------------------------------------------------
// Encoder: h0 = PQ @ W_enc + b_enc ; V_pred = [1,0] ; write f16 x row.
// ---------------------------------------------------------------------------
__global__ void encoder_kernel(const float* __restrict__ PQ,
                               const float* __restrict__ Wenc,
                               const float* __restrict__ benc,
                               float* __restrict__ V,
                               _Float16* __restrict__ x16, int N_)
{
  const int nd = blockIdx.x * blockDim.x + threadIdx.x;
  if (nd >= N_) return;
  const float p = PQ[2 * (size_t)nd], q = PQ[2 * (size_t)nd + 1];
  V[2 * (size_t)nd]     = 1.0f;
  V[2 * (size_t)nd + 1] = 0.0f;
  _Float16* row = x16 + (size_t)nd * XW;
  row[0] = (_Float16)1.0f;
  row[1] = (_Float16)0.0f;
  #pragma unroll
  for (int j = 0; j < HID; ++j)
    row[2 + j] = (_Float16)(benc[j] + p * Wenc[j] + q * Wenc[HID + j]);
  row[34] = (_Float16)0.0f;
  row[35] = (_Float16)0.0f;
}

// ---------------------------------------------------------------------------
// Edge messages: m = relu([x[s],x[r],ef] @ Wmsg + b) * mask ; atomic segment sum
// One 16-edge tile per wave; A staged in per-wave LDS; 6 WMMA per tile.
// ---------------------------------------------------------------------------
__global__ void edge_msg_kernel(const _Float16* __restrict__ x16,
                                const int* __restrict__ senders,
                                const int* __restrict__ receivers,
                                const float* __restrict__ efeat,
                                const float* __restrict__ emask,
                                const _Float16* __restrict__ wfrag,
                                const float* __restrict__ bias,
                                float* __restrict__ agg,
                                int E_, int ntiles)
{
  __shared__ _Float16 tiles[8][16 * KP];
  __shared__ int      rcv_s[8][16];
  const int wave = threadIdx.x >> 5, lane = threadIdx.x & 31;
  const int tile = blockIdx.x * 8 + wave;

  v16h B[3][2];
  #pragma unroll
  for (int c = 0; c < 3; ++c)
    #pragma unroll
    for (int t = 0; t < 2; ++t)
      B[c][t] = *(const v16h*)(wfrag + (size_t)((c * 2 + t) * 32 + lane) * 16);

  if (tile >= ntiles) return;

  _Float16* tm = tiles[wave];
  int*      rl = rcv_s[wave];
  const int e0       = tile * 16;
  const int e        = e0 + (lane >> 1);
  const bool fullTile = (e0 + 16) <= E_;
  unsigned* tU = (unsigned*)(tm + (size_t)(lane >> 1) * KP);

  if (fullTile) {                         // hot path: no per-element predication
    if ((lane & 1) == 0) {
      const unsigned* src = (const unsigned*)(x16 + (size_t)senders[e] * XW);
      #pragma unroll
      for (int i = 0; i < 17; ++i) tU[i] = src[i];             // cols 0..33
      union { _Float16 h[2]; unsigned u; } p;
      p.h[0] = (_Float16)efeat[2 * (size_t)e];
      p.h[1] = (_Float16)efeat[2 * (size_t)e + 1];
      tU[34] = p.u;                                            // cols 68..69
      #pragma unroll
      for (int i = 35; i < 41; ++i) tU[i] = 0u;
    } else {
      const int r = receivers[e];
      rl[lane >> 1] = r;
      const unsigned* src = (const unsigned*)(x16 + (size_t)r * XW);
      #pragma unroll
      for (int i = 0; i < 17; ++i) tU[17 + i] = src[i];        // cols 34..67
      #pragma unroll
      for (int i = 41; i < 48; ++i) tU[i] = 0u;
    }
  } else {                                // tail tile
    const bool valid = e < E_;
    if ((lane & 1) == 0) {
      const int s = valid ? senders[e] : 0;
      const unsigned* src = (const unsigned*)(x16 + (size_t)s * XW);
      #pragma unroll
      for (int i = 0; i < 17; ++i) tU[i] = valid ? src[i] : 0u;
      union { _Float16 h[2]; unsigned u; } p;
      p.h[0] = (_Float16)(valid ? efeat[2 * (size_t)e]     : 0.0f);
      p.h[1] = (_Float16)(valid ? efeat[2 * (size_t)e + 1] : 0.0f);
      tU[34] = p.u;
      #pragma unroll
      for (int i = 35; i < 41; ++i) tU[i] = 0u;
    } else {
      const int r = valid ? receivers[e] : 0;
      rl[lane >> 1] = r;
      const unsigned* src = (const unsigned*)(x16 + (size_t)r * XW);
      #pragma unroll
      for (int i = 0; i < 17; ++i) tU[17 + i] = valid ? src[i] : 0u;
      #pragma unroll
      for (int i = 41; i < 48; ++i) tU[i] = 0u;
    }
  }
  asm volatile("s_wait_dscnt 0" ::: "memory");  // per-wave LDS store->load order

  v8f acc[2] = {};
  #pragma unroll
  for (int c = 0; c < 3; ++c) {
    v16h a = load_a_frag(tm, lane, c, KP / 2);
    acc[0] = __builtin_amdgcn_wmma_f32_16x16x32_f16(false, a, false, B[c][0],
                                                    (short)0, acc[0], false, false);
    acc[1] = __builtin_amdgcn_wmma_f32_16x16x32_f16(false, a, false, B[c][1],
                                                    (short)0, acc[1], false, false);
  }

  const int ncol = lane & 15, rbase = (lane >> 4) * 8;
  const float b0 = bias[ncol];
  const float b1 = bias[ncol + 16];
  if (fullTile) {                         // branch-free scatter
    const int4   r0 = *(const int4*)(rl + rbase);
    const int4   r1 = *(const int4*)(rl + rbase + 4);
    const float4 m0 = *(const float4*)(emask + e0 + rbase);
    const float4 m1 = *(const float4*)(emask + e0 + rbase + 4);
    const int   rv[8] = {r0.x, r0.y, r0.z, r0.w, r1.x, r1.y, r1.z, r1.w};
    const float mv[8] = {m0.x, m0.y, m0.z, m0.w, m1.x, m1.y, m1.z, m1.w};
    #pragma unroll
    for (int v = 0; v < 8; ++v) {
      float* rowp = agg + (size_t)rv[v] * HID;
      float v0 = acc[0][v] + b0; v0 = v0 > 0.0f ? v0 : 0.0f;
      float v1 = acc[1][v] + b1; v1 = v1 > 0.0f ? v1 : 0.0f;
      atomicAdd(rowp + ncol,      v0 * mv[v]);
      atomicAdd(rowp + ncol + 16, v1 * mv[v]);
    }
  } else {
    #pragma unroll
    for (int v = 0; v < 8; ++v) {
      const int ee = e0 + rbase + v;
      if (ee < E_) {
        const float mk = emask[ee];
        float* rowp = agg + (size_t)rl[rbase + v] * HID;
        float v0 = acc[0][v] + b0; v0 = v0 > 0.0f ? v0 : 0.0f;
        float v1 = acc[1][v] + b1; v1 = v1 > 0.0f ? v1 : 0.0f;
        atomicAdd(rowp + ncol,      v0 * mk);
        atomicAdd(rowp + ncol + 16, v1 * mk);
      }
    }
  }
}

// ---------------------------------------------------------------------------
// Node update: h = relu([x,agg] @ Wnode + b) ; accumulate pooled sum.
// ---------------------------------------------------------------------------
__global__ void node_update_kernel(const _Float16* __restrict__ x16,
                                   const float* __restrict__ agg,
                                   const _Float16* __restrict__ wfrag,
                                   const float* __restrict__ bias,
                                   float* __restrict__ hbuf,
                                   float* __restrict__ pooled,
                                   int N_, int ntiles)
{
  __shared__ _Float16 tiles[8][16 * KP];
  __shared__ float pool_s[HID];
  const int wave = threadIdx.x >> 5, lane = threadIdx.x & 31;
  const int tile = blockIdx.x * 8 + wave;
  if (threadIdx.x < HID) pool_s[threadIdx.x] = 0.0f;
  __syncthreads();

  v16h B[3][2];
  #pragma unroll
  for (int c = 0; c < 3; ++c)
    #pragma unroll
    for (int t = 0; t < 2; ++t)
      B[c][t] = *(const v16h*)(wfrag + (size_t)((c * 2 + t) * 32 + lane) * 16);

  if (tile < ntiles) {
    _Float16* tm = tiles[wave];
    const int nd        = tile * 16 + (lane >> 1);
    const bool fullTile = (tile * 16 + 16) <= N_;
    const bool valid    = nd < N_;
    unsigned* tU = (unsigned*)(tm + (size_t)(lane >> 1) * KP);
    if ((lane & 1) == 0) {
      const unsigned* src = (const unsigned*)(x16 + (size_t)(valid ? nd : 0) * XW);
      #pragma unroll
      for (int i = 0; i < 17; ++i) tU[i] = valid ? src[i] : 0u;  // x cols 0..33
      #pragma unroll
      for (int i = 33; i < 48; ++i) tU[i] = 0u;                  // pad 66..95
    } else {
      const float* ar = agg + (size_t)(valid ? nd : 0) * HID;
      #pragma unroll
      for (int i = 0; i < 16; ++i) {                             // agg cols 34..65
        union { _Float16 h[2]; unsigned u; } p;
        p.h[0] = (_Float16)(valid ? ar[2 * i]     : 0.0f);
        p.h[1] = (_Float16)(valid ? ar[2 * i + 1] : 0.0f);
        tU[17 + i] = p.u;
      }
    }
    asm volatile("s_wait_dscnt 0" ::: "memory");

    v8f acc[2] = {};
    #pragma unroll
    for (int c = 0; c < 3; ++c) {
      v16h a = load_a_frag(tm, lane, c, KP / 2);
      acc[0] = __builtin_amdgcn_wmma_f32_16x16x32_f16(false, a, false, B[c][0],
                                                      (short)0, acc[0], false, false);
      acc[1] = __builtin_amdgcn_wmma_f32_16x16x32_f16(false, a, false, B[c][1],
                                                      (short)0, acc[1], false, false);
    }
    const int ncol = lane & 15, rbase = (lane >> 4) * 8;
    #pragma unroll
    for (int t = 0; t < 2; ++t) {
      const int n = ncol + t * 16;
      const float b = bias[n];
      float part = 0.0f;
      if (fullTile) {
        #pragma unroll
        for (int v = 0; v < 8; ++v) {
          const int node = tile * 16 + rbase + v;
          float val = acc[t][v] + b;
          val = val > 0.0f ? val : 0.0f;
          hbuf[(size_t)node * HID + n] = val;
          part += val;
        }
      } else {
        #pragma unroll
        for (int v = 0; v < 8; ++v) {
          const int node = tile * 16 + rbase + v;
          if (node < N_) {
            float val = acc[t][v] + b;
            val = val > 0.0f ? val : 0.0f;
            hbuf[(size_t)node * HID + n] = val;
            part += val;
          }
        }
      }
      atomicAdd(&pool_s[n], part);      // LDS atomic
    }
  }
  __syncthreads();
  if (threadIdx.x < HID) atomicAdd(&pooled[threadIdx.x], pool_s[threadIdx.x]);
}

// ---------------------------------------------------------------------------
// Global super-node update (1 wave): g = relu([g, pooled/N] @ Wg + bg)
// ---------------------------------------------------------------------------
__global__ void g_update_kernel(const float* __restrict__ pooled,
                                const float* __restrict__ Wg,
                                const float* __restrict__ bg,
                                float* __restrict__ g,
                                _Float16* __restrict__ g16, float invN)
{
  const int n = threadIdx.x;   // 32 threads
  float gin[64];
  #pragma unroll
  for (int k = 0; k < HID; ++k) gin[k] = g[k];
  #pragma unroll
  for (int k = 0; k < HID; ++k) gin[HID + k] = pooled[k] * invN;
  __syncthreads();
  float acc = bg[n];
  #pragma unroll
  for (int k = 0; k < 64; ++k) acc += gin[k] * Wg[k * HID + n];
  acc = acc > 0.0f ? acc : 0.0f;
  g[n]   = acc;
  g16[n] = (_Float16)acc;
}

// ---------------------------------------------------------------------------
// Broadcast: h = relu([h, g] @ Wgn + b) ; rewrite h (in place) and x16[:,2:34]
// ---------------------------------------------------------------------------
__global__ void global_node_kernel(float* __restrict__ hbuf,
                                   const _Float16* __restrict__ g16,
                                   const _Float16* __restrict__ wfrag,
                                   const float* __restrict__ bias,
                                   _Float16* __restrict__ x16,
                                   int N_, int ntiles)
{
  __shared__ _Float16 tiles[8][16 * 64];
  const int wave = threadIdx.x >> 5, lane = threadIdx.x & 31;
  const int tile = blockIdx.x * 8 + wave;

  v16h B[2][2];
  #pragma unroll
  for (int c = 0; c < 2; ++c)
    #pragma unroll
    for (int t = 0; t < 2; ++t)
      B[c][t] = *(const v16h*)(wfrag + (size_t)((c * 2 + t) * 32 + lane) * 16);

  if (tile >= ntiles) return;

  _Float16* tm = tiles[wave];
  const int nd        = tile * 16 + (lane >> 1);
  const bool fullTile = (tile * 16 + 16) <= N_;
  const bool valid    = nd < N_;
  unsigned* tU = (unsigned*)(tm + (size_t)(lane >> 1) * 64);
  if ((lane & 1) == 0) {
    const float* hr = hbuf + (size_t)(valid ? nd : 0) * HID;
    #pragma unroll
    for (int i = 0; i < 16; ++i) {                 // h cols 0..31
      union { _Float16 h[2]; unsigned u; } p;
      p.h[0] = (_Float16)(valid ? hr[2 * i]     : 0.0f);
      p.h[1] = (_Float16)(valid ? hr[2 * i + 1] : 0.0f);
      tU[i] = p.u;
    }
  } else {
    const unsigned* gs = (const unsigned*)g16;     // g cols 32..63 (broadcast)
    #pragma unroll
    for (int i = 0; i < 16; ++i) tU[16 + i] = gs[i];
  }
  asm volatile("s_wait_dscnt 0" ::: "memory");

  v8f acc[2] = {};
  #pragma unroll
  for (int c = 0; c < 2; ++c) {
    v16h a = load_a_frag(tm, lane, c, 32);
    acc[0] = __builtin_amdgcn_wmma_f32_16x16x32_f16(false, a, false, B[c][0],
                                                    (short)0, acc[0], false, false);
    acc[1] = __builtin_amdgcn_wmma_f32_16x16x32_f16(false, a, false, B[c][1],
                                                    (short)0, acc[1], false, false);
  }
  const int ncol = lane & 15, rbase = (lane >> 4) * 8;
  #pragma unroll
  for (int t = 0; t < 2; ++t) {
    const int n = ncol + t * 16;
    const float b = bias[n];
    if (fullTile) {
      #pragma unroll
      for (int v = 0; v < 8; ++v) {
        const int node = tile * 16 + rbase + v;
        float val = acc[t][v] + b;
        val = val > 0.0f ? val : 0.0f;
        hbuf[(size_t)node * HID + n]   = val;
        x16[(size_t)node * XW + 2 + n] = (_Float16)val;
      }
    } else {
      #pragma unroll
      for (int v = 0; v < 8; ++v) {
        const int node = tile * 16 + rbase + v;
        if (node < N_) {
          float val = acc[t][v] + b;
          val = val > 0.0f ? val : 0.0f;
          hbuf[(size_t)node * HID + n]   = val;
          x16[(size_t)node * XW + 2 + n] = (_Float16)val;
        }
      }
    }
  }
}

// ---------------------------------------------------------------------------
// Voltage correction: V += h @ Wdv + bdv ; refresh x16[:,0:2]
// ---------------------------------------------------------------------------
__global__ void dv_kernel(const float* __restrict__ hbuf,
                          const float* __restrict__ Wdv,
                          const float* __restrict__ bdv,
                          float* __restrict__ V,
                          _Float16* __restrict__ x16, int N_)
{
  const int nd = blockIdx.x * blockDim.x + threadIdx.x;
  if (nd >= N_) return;
  const float* hr = hbuf + (size_t)nd * HID;
  float d0 = bdv[0], d1 = bdv[1];
  #pragma unroll
  for (int k = 0; k < HID; ++k) {
    const float hv = hr[k];
    d0 += hv * Wdv[2 * k];
    d1 += hv * Wdv[2 * k + 1];
  }
  const float v0 = V[2 * (size_t)nd]     + d0;
  const float v1 = V[2 * (size_t)nd + 1] + d1;
  V[2 * (size_t)nd]     = v0;
  V[2 * (size_t)nd + 1] = v1;
  x16[(size_t)nd * XW + 0] = (_Float16)v0;
  x16[(size_t)nd * XW + 1] = (_Float16)v1;
}

// ---------------------------------------------------------------------------
extern "C" void kernel_launch(void* const* d_in, const int* in_sizes, int n_in,
                              void* d_out, int out_size, void* d_ws, size_t ws_size,
                              hipStream_t stream)
{
  (void)n_in; (void)out_size; (void)ws_size;
  const float* PQ    = (const float*)d_in[0];
  const int*   snd   = (const int*)d_in[1];
  const int*   rcv   = (const int*)d_in[2];
  const float* efeat = (const float*)d_in[3];
  const float* emask = (const float*)d_in[4];
  const float* Wenc  = (const float*)d_in[5];
  const float* benc  = (const float*)d_in[6];
  const float* Wmsg  = (const float*)d_in[7];
  const float* bmsg  = (const float*)d_in[8];
  const float* Wnode = (const float*)d_in[9];
  const float* bnode = (const float*)d_in[10];
  const float* Wg    = (const float*)d_in[11];
  const float* bg    = (const float*)d_in[12];
  const float* Wgn   = (const float*)d_in[13];
  const float* bgn   = (const float*)d_in[14];
  const float* Wdv   = (const float*)d_in[15];
  const float* bdv   = (const float*)d_in[16];

  const int N = in_sizes[0] / 2;
  const int E = in_sizes[1];
  float* V = (float*)d_out;

  char* ws = (char*)d_ws;
  size_t off = 0;
  auto take = [&](size_t bytes) -> char* {
    char* p = ws + off;
    off = (off + bytes + 255) & ~(size_t)255;
    return p;
  };
  _Float16* x16    = (_Float16*)take((size_t)N * XW * 2);
  float*    hbuf   = (float*)take((size_t)N * HID * 4);
  float*    agg    = (float*)take(((size_t)N * HID + HID) * 4);
  float*    pooled = agg + (size_t)N * HID;            // contiguous for one memset
  float*    g      = (float*)take(HID * 4);
  _Float16* g16    = (_Float16*)take(HID * 2);
  _Float16* fmsg   = (_Float16*)take((size_t)LAYERS * 6 * 512 * 2);
  _Float16* fnode  = (_Float16*)take((size_t)LAYERS * 6 * 512 * 2);
  _Float16* fgn    = (_Float16*)take((size_t)LAYERS * 4 * 512 * 2);

  hipMemsetAsync(g, 0, HID * 4, stream);
  prep_weights<<<dim3(16), dim3(256), 0, stream>>>(Wmsg, Wnode, Wgn, fmsg, fnode, fgn);
  encoder_kernel<<<dim3((N + 255) / 256), dim3(256), 0, stream>>>(PQ, Wenc, benc, V, x16, N);

  const int ntE = (E + 15) / 16, gE = (ntE + 7) / 8;
  const int ntN = (N + 15) / 16, gN = (ntN + 7) / 8;
  for (int l = 0; l < LAYERS; ++l) {
    hipMemsetAsync(agg, 0, ((size_t)N * HID + HID) * 4, stream);
    edge_msg_kernel<<<dim3(gE), dim3(256), 0, stream>>>(
        x16, snd, rcv, efeat, emask,
        fmsg + (size_t)l * 6 * 512, bmsg + l * HID, agg, E, ntE);
    node_update_kernel<<<dim3(gN), dim3(256), 0, stream>>>(
        x16, agg, fnode + (size_t)l * 6 * 512, bnode + l * HID,
        hbuf, pooled, N, ntN);
    g_update_kernel<<<dim3(1), dim3(32), 0, stream>>>(
        pooled, Wg + (size_t)l * 64 * HID, bg + l * HID, g, g16, 1.0f / (float)N);
    global_node_kernel<<<dim3(gN), dim3(256), 0, stream>>>(
        hbuf, g16, fgn + (size_t)l * 4 * 512, bgn + l * HID, x16, N, ntN);
    dv_kernel<<<dim3((N + 255) / 256), dim3(256), 0, stream>>>(
        hbuf, Wdv + (size_t)l * HID * 2, bdv + l * 2, V, x16, N);
  }
}
